// QueryInteractionModule_47090021433845
// MI455X (gfx1250) — matrix operand order repaired and checked by reading, not compiled, not run on hardware
//
#include <hip/hip_runtime.h>
#include <hip/hip_bf16.h>
#include <math.h>

// ---- problem constants (from reference) ----
#define L_SEQ 8192      // mamba sequence length (2*M)
#define DM    256       // D_MODEL
#define DIN   512       // D_IN
#define HIDD  1024      // HID
#define MQ    4096      // M
#define NST   16        // N_STATE
#define NCH   64        // scan chunks
#define CLEN  128       // scan chunk length (NCH*CLEN == L_SEQ)

typedef __bf16 bf16;
typedef __attribute__((ext_vector_type(16))) __bf16 v16bf;
typedef __attribute__((ext_vector_type(8)))  __bf16 v8bf;
typedef __attribute__((ext_vector_type(8)))  float  v8f;

// ---------------------------------------------------------------------------
// Build x = concat(oe[:2048], oe+qpe, oe[2048:]), store residual, RMSNorm->bf16
// ---------------------------------------------------------------------------
__global__ void build_h_kernel(const float* __restrict__ oe, const float* __restrict__ qpe,
                               const float* __restrict__ rmsw,
                               float* __restrict__ resid, bf16* __restrict__ hbf)
{
    int l = blockIdx.x;
    int d = threadIdx.x;
    float v;
    if (l < 2048)        v = oe[(size_t)l * DM + d];
    else if (l < 6144) { int r = l - 2048; v = oe[(size_t)r * DM + d] + qpe[(size_t)r * DM + d]; }
    else               { int r = l - 4096; v = oe[(size_t)r * DM + d]; }

    __shared__ float red[DM];
    red[d] = v * v;
    __syncthreads();
    for (int s = DM / 2; s > 0; s >>= 1) {
        if (d < s) red[d] += red[d + s];
        __syncthreads();
    }
    float scale = rsqrtf(red[0] * (1.0f / DM) + 1e-5f);
    resid[(size_t)l * DM + d] = v;
    hbf[(size_t)l * DM + d]   = (bf16)(v * scale * rmsw[d]);
}

// ---------------------------------------------------------------------------
// LayerNorm over 256 cols: out = LN(inA (+ inB)) * w + b ; optional bf16 copy
// ---------------------------------------------------------------------------
__global__ void ln_kernel(const float* __restrict__ inA, const float* __restrict__ inB,
                          const float* __restrict__ w, const float* __restrict__ b,
                          float* __restrict__ outF, bf16* __restrict__ outB)
{
    int r = blockIdx.x, d = threadIdx.x;
    float v = inA[(size_t)r * DM + d];
    if (inB) v += inB[(size_t)r * DM + d];
    __shared__ float s1[DM], s2[DM];
    s1[d] = v; s2[d] = v * v;
    __syncthreads();
    for (int s = DM / 2; s > 0; s >>= 1) {
        if (d < s) { s1[d] += s1[d + s]; s2[d] += s2[d + s]; }
        __syncthreads();
    }
    float mu  = s1[0] * (1.0f / DM);
    float var = s2[0] * (1.0f / DM) - mu * mu;
    float o = (v - mu) * rsqrtf(var + 1e-5f) * w[d] + b[d];
    outF[(size_t)r * DM + d] = o;
    if (outB) outB[(size_t)r * DM + d] = (bf16)o;
}

// ---------------------------------------------------------------------------
__global__ void cvt_bf16_kernel(const float* __restrict__ in, bf16* __restrict__ out, int n)
{
    int i = blockIdx.x * blockDim.x + threadIdx.x;
    if (i < n) out[i] = (bf16)in[i];
}

// ---------------------------------------------------------------------------
// bf16 WMMA GEMM: C[M,N] = A[M,K] @ W[N,K]^T  (+bias)(+res)(relu)
// Wave computes a 16 x (NT*16) tile; block = 8 waves -> 128 x (NT*16) tile.
// A fragment layout (16-bit A 16x32): lane m = lane&15; K runs of 8 at
//   k0 + (lane>>4)*8 and k0 + 16 + (lane>>4)*8.
// B fragment layout (16-bit B 32x16): lane n = lane&15; K runs of 8 at
//   k0 + (lane>>4)*8 and k0 + 16 + (lane>>4)*8.
// C/D layout: lane -> N, vgpr v -> M = v + 8*(lane>>4).
// ---------------------------------------------------------------------------
template<int NT, bool HAS_BIAS, bool RELU, bool ADD_RES, bool OUT_F32, bool OUT_BF16>
__global__ void gemm_wmma_kernel(const bf16* __restrict__ A, const bf16* __restrict__ W,
                                 const float* __restrict__ bias, const float* __restrict__ res,
                                 float* __restrict__ outF, bf16* __restrict__ outB,
                                 int N, int K)
{
    const int wave = threadIdx.x >> 5;
    const int lane = threadIdx.x & 31;
    const int lr = lane & 15;
    const int lh = lane >> 4;
    const int m0 = (blockIdx.x * 8 + wave) * 16;
    const int n0 = blockIdx.y * (NT * 16);

    v8f acc[NT] = {};
    const bf16* Arow = A + (size_t)(m0 + lr) * K;

    for (int k0 = 0; k0 < K; k0 += 32) {
        v8bf alo = *(const v8bf*)(Arow + k0 + lh * 8);
        v8bf ahi = *(const v8bf*)(Arow + k0 + 16 + lh * 8);
        v16bf afrag = __builtin_shufflevector(alo, ahi,
            0,1,2,3,4,5,6,7,8,9,10,11,12,13,14,15);
#pragma unroll
        for (int t = 0; t < NT; ++t) {
            const bf16* Wrow = W + (size_t)(n0 + t * 16 + lr) * K;
            v8bf blo = *(const v8bf*)(Wrow + k0 + lh * 8);
            v8bf bhi = *(const v8bf*)(Wrow + k0 + 16 + lh * 8);
            v16bf bfrag = __builtin_shufflevector(blo, bhi,
                0,1,2,3,4,5,6,7,8,9,10,11,12,13,14,15);
            acc[t] = __builtin_amdgcn_wmma_f32_16x16x32_bf16(
                false, afrag, false, bfrag, (short)0, acc[t], false, false);
        }
    }

#pragma unroll
    for (int t = 0; t < NT; ++t) {
        int n = n0 + t * 16 + lr;
        float bv = HAS_BIAS ? bias[n] : 0.0f;
#pragma unroll
        for (int v = 0; v < 8; ++v) {
            int m = m0 + v + 8 * lh;
            float val = acc[t][v] + bv;
            if (ADD_RES) val += res[(size_t)m * N + n];
            if (RELU)    val = fmaxf(val, 0.0f);
            if (OUT_F32) outF[(size_t)m * N + n] = val;
            if (OUT_BF16) outB[(size_t)m * N + n] = (bf16)val;
        }
    }
}

// ---------------------------------------------------------------------------
// depthwise causal conv (k=4) + SiLU, directional read of x-part of xz
// ---------------------------------------------------------------------------
__global__ void conv_silu_kernel(const float* __restrict__ xz, const float* __restrict__ w,
                                 const float* __restrict__ bias, float* __restrict__ xcF,
                                 bf16* __restrict__ xcB, int dir)
{
    int gid = blockIdx.x * blockDim.x + threadIdx.x;
    int tau = gid >> 9;
    int d   = gid & 511;
    float acc = bias[d];
#pragma unroll
    for (int k = 0; k < 4; ++k) {
        int t = tau - 3 + k;
        if (t >= 0) {
            int l = dir ? (L_SEQ - 1 - t) : t;
            acc += w[d * 4 + k] * xz[(size_t)l * (2 * DIN) + d];
        }
    }
    float s = acc * (1.0f / (1.0f + __expf(-acc)));
    xcF[(size_t)tau * DIN + d] = s;
    xcB[(size_t)tau * DIN + d] = (bf16)s;
}

// ---------------------------------------------------------------------------
// delta = softplus(dt @ dt_proj_w^T + dt_proj_b); dt = xdbl[:, :16]
// ---------------------------------------------------------------------------
__global__ void delta_kernel(const float* __restrict__ xdbl, const float* __restrict__ dtw,
                             const float* __restrict__ dtb, float* __restrict__ delta)
{
    int gid = blockIdx.x * blockDim.x + threadIdx.x;
    int tau = gid >> 9;
    int d   = gid & 511;
    const float* xr = xdbl + (size_t)tau * 48;
    float acc = dtb[d];
#pragma unroll
    for (int r = 0; r < 16; ++r) acc += xr[r] * dtw[d * 16 + r];
    delta[(size_t)tau * DIN + d] = (acc > 20.0f) ? acc : log1pf(__expf(acc));
}

// ---------------------------------------------------------------------------
// Chunked parallel scan, pass 1: per (chunk,d) compute prod(a) and local h_end
// ---------------------------------------------------------------------------
__global__ void scan_pass1_kernel(const float* __restrict__ delta, const float* __restrict__ xc,
                                  const float* __restrict__ xdbl, const float* __restrict__ Alog,
                                  float* __restrict__ Aprod, float* __restrict__ Hend)
{
    int gid = blockIdx.x * blockDim.x + threadIdx.x;
    int c = gid >> 9;
    int d = gid & 511;
    float Aval[NST], hp[NST], ap[NST];
#pragma unroll
    for (int n = 0; n < NST; ++n) { Aval[n] = -__expf(Alog[d * NST + n]); hp[n] = 0.f; ap[n] = 1.f; }
    int base = c * CLEN;
    for (int i = 0; i < CLEN; ++i) {
        int tau = base + i;
        float dv = delta[(size_t)tau * DIN + d];
        float xv = xc[(size_t)tau * DIN + d];
        const float* B = xdbl + (size_t)tau * 48 + 16;
#pragma unroll
        for (int n = 0; n < NST; ++n) {
            float a = __expf(dv * Aval[n]);
            hp[n] = a * hp[n] + dv * B[n] * xv;
            ap[n] *= a;
        }
    }
    size_t o = ((size_t)c * DIN + d) * NST;
#pragma unroll
    for (int n = 0; n < NST; ++n) { Aprod[o + n] = ap[n]; Hend[o + n] = hp[n]; }
}

// pass 2: tiny serial sweep over chunks per (d,n) -> seed states
__global__ void scan_pass2_kernel(const float* __restrict__ Aprod, const float* __restrict__ Hend,
                                  float* __restrict__ Hin)
{
    int gid = blockIdx.x * blockDim.x + threadIdx.x;   // gid = d*16+n
    if (gid >= DIN * NST) return;
    float h = 0.f;
    for (int c = 0; c < NCH; ++c) {
        size_t idx = (size_t)c * DIN * NST + gid;
        Hin[idx] = h;
        h = Aprod[idx] * h + Hend[idx];
    }
}

// pass 3: seeded replay producing y = h@C + D*xc, gated by silu(z); dir merge
__global__ void scan_pass3_kernel(const float* __restrict__ delta, const float* __restrict__ xc,
                                  const float* __restrict__ xdbl, const float* __restrict__ Alog,
                                  const float* __restrict__ Dp, const float* __restrict__ Hin,
                                  const float* __restrict__ xz,
                                  float* __restrict__ ycomb, bf16* __restrict__ ycombBf, int dir)
{
    int gid = blockIdx.x * blockDim.x + threadIdx.x;
    int c = gid >> 9;
    int d = gid & 511;
    float Aval[NST], h[NST];
    size_t ho = ((size_t)c * DIN + d) * NST;
#pragma unroll
    for (int n = 0; n < NST; ++n) { Aval[n] = -__expf(Alog[d * NST + n]); h[n] = Hin[ho + n]; }
    float Dv = Dp[d];
    int base = c * CLEN;
    for (int i = 0; i < CLEN; ++i) {
        int tau = base + i;
        float dv = delta[(size_t)tau * DIN + d];
        float xv = xc[(size_t)tau * DIN + d];
        const float* B = xdbl + (size_t)tau * 48 + 16;
        const float* C = xdbl + (size_t)tau * 48 + 32;
        float y = 0.f;
#pragma unroll
        for (int n = 0; n < NST; ++n) {
            float a = __expf(dv * Aval[n]);
            h[n] = a * h[n] + dv * B[n] * xv;
            y += h[n] * C[n];
        }
        y += Dv * xv;
        int l = dir ? (L_SEQ - 1 - tau) : tau;
        float z = xz[(size_t)l * (2 * DIN) + DIN + d];
        float out = y * (z * (1.0f / (1.0f + __expf(-z))));
        if (dir == 0) {
            ycomb[(size_t)l * DIN + d] = out;
        } else {
            float comb = 0.5f * (ycomb[(size_t)l * DIN + d] + out);
            ycomb[(size_t)l * DIN + d]  = comb;
            ycombBf[(size_t)l * DIN + d] = (bf16)comb;
        }
    }
}

// ---------------------------------------------------------------------------
extern "C" void kernel_launch(void* const* d_in, const int* in_sizes, int n_in,
                              void* d_out, int out_size, void* d_ws, size_t ws_size,
                              hipStream_t stream)
{
    (void)in_sizes; (void)n_in; (void)out_size; (void)ws_size;

    const float* oe    = (const float*)d_in[0];
    const float* qpe   = (const float*)d_in[1];
    const float* qft   = (const float*)d_in[2];
    const float* in_proj_w  = (const float*)d_in[3];
    const float* conv_w_f   = (const float*)d_in[4];
    const float* conv_b_f   = (const float*)d_in[5];
    const float* x_proj_w_f = (const float*)d_in[6];
    const float* dt_w_f     = (const float*)d_in[7];
    const float* dt_b_f     = (const float*)d_in[8];
    const float* Alog_f     = (const float*)d_in[9];
    const float* Dp_f       = (const float*)d_in[10];
    const float* conv_w_b   = (const float*)d_in[11];
    const float* conv_b_b   = (const float*)d_in[12];
    const float* x_proj_w_b = (const float*)d_in[13];
    const float* dt_w_b     = (const float*)d_in[14];
    const float* dt_b_b     = (const float*)d_in[15];
    const float* Alog_b     = (const float*)d_in[16];
    const float* Dp_b       = (const float*)d_in[17];
    const float* out_proj_w = (const float*)d_in[18];
    const float* rms_w      = (const float*)d_in[19];
    const float* l1w  = (const float*)d_in[20];
    const float* l1b  = (const float*)d_in[21];
    const float* lp1w = (const float*)d_in[22];
    const float* lp1b = (const float*)d_in[23];
    const float* lf1w = (const float*)d_in[24];
    const float* lf1b = (const float*)d_in[25];
    const float* l2w  = (const float*)d_in[26];
    const float* l2b  = (const float*)d_in[27];
    const float* lp2w = (const float*)d_in[28];
    const float* lp2b = (const float*)d_in[29];
    const float* lf2w = (const float*)d_in[30];
    const float* lf2b = (const float*)d_in[31];
    const float* n1w = (const float*)d_in[32];
    const float* n1b = (const float*)d_in[33];
    const float* n2w = (const float*)d_in[34];
    const float* n2b = (const float*)d_in[35];
    const float* npw = (const float*)d_in[36];
    const float* npb = (const float*)d_in[37];
    const float* nfw = (const float*)d_in[38];
    const float* nfb = (const float*)d_in[39];
    float* out = (float*)d_out;

    // ---- workspace arena ----
    char* p = (char*)d_ws;
    auto carve = [&](size_t bytes) -> char* {
        char* r = p; p += (bytes + 255) & ~(size_t)255; return r;
    };
    float* resid    = (float*)carve((size_t)L_SEQ * DM * 4);
    bf16*  hbf      = (bf16*) carve((size_t)L_SEQ * DM * 2);
    bf16*  w_inproj = (bf16*) carve((size_t)2 * DIN * DM * 2);
    bf16*  w_outproj= (bf16*) carve((size_t)DM * DIN * 2);
    bf16*  w_xp_f   = (bf16*) carve((size_t)48 * DIN * 2);
    bf16*  w_xp_b   = (bf16*) carve((size_t)48 * DIN * 2);
    bf16*  w_l1     = (bf16*) carve((size_t)HIDD * DM * 2);
    bf16*  w_lp1    = (bf16*) carve((size_t)HIDD * DM * 2);
    bf16*  w_lf1    = (bf16*) carve((size_t)HIDD * DM * 2);
    bf16*  w_l2     = (bf16*) carve((size_t)DM * HIDD * 2);
    bf16*  w_lp2    = (bf16*) carve((size_t)DM * HIDD * 2);
    bf16*  w_lf2    = (bf16*) carve((size_t)DM * HIDD * 2);
    float* xz       = (float*)carve((size_t)L_SEQ * 2 * DIN * 4);
    float* xcF_f    = (float*)carve((size_t)L_SEQ * DIN * 4);
    bf16*  xcB_f    = (bf16*) carve((size_t)L_SEQ * DIN * 2);
    float* xcF_b    = (float*)carve((size_t)L_SEQ * DIN * 4);
    bf16*  xcB_b    = (bf16*) carve((size_t)L_SEQ * DIN * 2);
    float* xdbl_f   = (float*)carve((size_t)L_SEQ * 48 * 4);
    float* xdbl_b   = (float*)carve((size_t)L_SEQ * 48 * 4);
    float* delta_f  = (float*)carve((size_t)L_SEQ * DIN * 4);
    float* delta_b  = (float*)carve((size_t)L_SEQ * DIN * 4);
    float* Aprod_f  = (float*)carve((size_t)NCH * DIN * NST * 4);
    float* Hend_f   = (float*)carve((size_t)NCH * DIN * NST * 4);
    float* Hin_f    = (float*)carve((size_t)NCH * DIN * NST * 4);
    float* Aprod_b  = (float*)carve((size_t)NCH * DIN * NST * 4);
    float* Hend_b   = (float*)carve((size_t)NCH * DIN * NST * 4);
    float* Hin_b    = (float*)carve((size_t)NCH * DIN * NST * 4);
    float* ycomb    = (float*)carve((size_t)L_SEQ * DIN * 4);
    bf16*  ycombBf  = (bf16*) carve((size_t)L_SEQ * DIN * 2);
    float* x2       = (float*)carve((size_t)L_SEQ * DM * 4);
    float* tgtA     = (float*)carve((size_t)MQ * DM * 4);
    bf16*  tgtA_bf  = (bf16*) carve((size_t)MQ * DM * 2);
    float* tgtB     = (float*)carve((size_t)MQ * DM * 4);
    bf16*  tgtB_bf  = (bf16*) carve((size_t)MQ * DM * 2);
    bf16*  hid_bf   = (bf16*) carve((size_t)MQ * HIDD * 2);
    float* preln    = (float*)carve((size_t)MQ * DM * 4);
    float* tmp256   = (float*)carve((size_t)MQ * DM * 4);

    auto cvt = [&](const float* src, bf16* dst, int n) {
        cvt_bf16_kernel<<<(n + 255) / 256, 256, 0, stream>>>(src, dst, n);
    };

    // 1. sequence assembly + RMSNorm
    build_h_kernel<<<L_SEQ, 256, 0, stream>>>(oe, qpe, rms_w, resid, hbf);

    // 2. weight conversion to bf16
    cvt(in_proj_w,  w_inproj,  2 * DIN * DM);
    cvt(out_proj_w, w_outproj, DM * DIN);
    cvt(x_proj_w_f, w_xp_f,    48 * DIN);
    cvt(x_proj_w_b, w_xp_b,    48 * DIN);
    cvt(l1w,  w_l1,  HIDD * DM);
    cvt(lp1w, w_lp1, HIDD * DM);
    cvt(lf1w, w_lf1, HIDD * DM);
    cvt(l2w,  w_l2,  DM * HIDD);
    cvt(lp2w, w_lp2, DM * HIDD);
    cvt(lf2w, w_lf2, DM * HIDD);

    // 3. in_proj: [8192,256] x [1024,256]^T -> xz [8192,1024]
    dim3 gInProj(L_SEQ / 128, (2 * DIN) / 64);
    gemm_wmma_kernel<4, false, false, false, true, false><<<gInProj, 256, 0, stream>>>(
        hbf, w_inproj, nullptr, nullptr, xz, nullptr, 2 * DIN, DM);

    // 4. causal conv + SiLU (both directions)
    int gConv = (L_SEQ * DIN) / 256;
    conv_silu_kernel<<<gConv, 256, 0, stream>>>(xz, conv_w_f, conv_b_f, xcF_f, xcB_f, 0);
    conv_silu_kernel<<<gConv, 256, 0, stream>>>(xz, conv_w_b, conv_b_b, xcF_b, xcB_b, 1);

    // 5. x_proj: [8192,512] x [48,512]^T -> x_dbl [8192,48]
    dim3 gXp(L_SEQ / 128, 1);
    gemm_wmma_kernel<3, false, false, false, true, false><<<gXp, 256, 0, stream>>>(
        xcB_f, w_xp_f, nullptr, nullptr, xdbl_f, nullptr, 48, DIN);
    gemm_wmma_kernel<3, false, false, false, true, false><<<gXp, 256, 0, stream>>>(
        xcB_b, w_xp_b, nullptr, nullptr, xdbl_b, nullptr, 48, DIN);

    // 6. delta = softplus(dt_proj)
    delta_kernel<<<gConv, 256, 0, stream>>>(xdbl_f, dt_w_f, dt_b_f, delta_f);
    delta_kernel<<<gConv, 256, 0, stream>>>(xdbl_b, dt_w_b, dt_b_b, delta_b);

    // 7-9. chunked parallel selective scan
    int gScan = (NCH * DIN) / 256;
    scan_pass1_kernel<<<gScan, 256, 0, stream>>>(delta_f, xcF_f, xdbl_f, Alog_f, Aprod_f, Hend_f);
    scan_pass1_kernel<<<gScan, 256, 0, stream>>>(delta_b, xcF_b, xdbl_b, Alog_b, Aprod_b, Hend_b);
    scan_pass2_kernel<<<(DIN * NST) / 256, 256, 0, stream>>>(Aprod_f, Hend_f, Hin_f);
    scan_pass2_kernel<<<(DIN * NST) / 256, 256, 0, stream>>>(Aprod_b, Hend_b, Hin_b);
    scan_pass3_kernel<<<gScan, 256, 0, stream>>>(delta_f, xcF_f, xdbl_f, Alog_f, Dp_f, Hin_f,
                                                 xz, ycomb, ycombBf, 0);
    scan_pass3_kernel<<<gScan, 256, 0, stream>>>(delta_b, xcF_b, xdbl_b, Alog_b, Dp_b, Hin_b,
                                                 xz, ycomb, ycombBf, 1);

    // 10. out_proj + residual -> x2 [8192,256]
    dim3 gOutProj(L_SEQ / 128, DM / 64);
    gemm_wmma_kernel<4, false, false, true, true, false><<<gOutProj, 256, 0, stream>>>(
        ycombBf, w_outproj, nullptr, resid, x2, nullptr, DM, DIN);

    // 11. tgt = LN(out_embed + x2[2048:6144])  (norm1)
    ln_kernel<<<MQ, 256, 0, stream>>>(oe, x2 + (size_t)2048 * DM, n1w, n1b, tgtA, tgtA_bf);

    // 12. FFN: hid = relu(tgt@l1^T + b1); preln = hid@l2^T + b2 + tgt
    dim3 gF1(MQ / 128, HIDD / 64);
    dim3 gF2(MQ / 128, DM / 64);
    gemm_wmma_kernel<4, true, true, false, false, true><<<gF1, 256, 0, stream>>>(
        tgtA_bf, w_l1, l1b, nullptr, nullptr, hid_bf, HIDD, DM);
    gemm_wmma_kernel<4, true, false, true, true, false><<<gF2, 256, 0, stream>>>(
        hid_bf, w_l2, l2b, tgtA, preln, nullptr, DM, HIDD);
    ln_kernel<<<MQ, 256, 0, stream>>>(preln, nullptr, n2w, n2b, tgtB, tgtB_bf);

    // 13. pos branch -> q_pos at out + MQ*DM
    gemm_wmma_kernel<4, true, true, false, false, true><<<gF1, 256, 0, stream>>>(
        tgtB_bf, w_lp1, lp1b, nullptr, nullptr, hid_bf, HIDD, DM);
    gemm_wmma_kernel<4, true, false, false, true, false><<<gF2, 256, 0, stream>>>(
        hid_bf, w_lp2, lp2b, nullptr, tmp256, nullptr, DM, HIDD);
    ln_kernel<<<MQ, 256, 0, stream>>>(qpe, tmp256, npw, npb, out + (size_t)MQ * DM, nullptr);

    // 14. feat branch -> q_feat at out
    gemm_wmma_kernel<4, true, true, false, false, true><<<gF1, 256, 0, stream>>>(
        tgtB_bf, w_lf1, lf1b, nullptr, nullptr, hid_bf, HIDD, DM);
    gemm_wmma_kernel<4, true, false, false, true, false><<<gF2, 256, 0, stream>>>(
        hid_bf, w_lf2, lf2b, nullptr, tmp256, nullptr, DM, HIDD);
    ln_kernel<<<MQ, 256, 0, stream>>>(qft, tmp256, nfw, nfb, out, nullptr);
}
